// GAT_85950885528277
// MI455X (gfx1250) — matrix operand order repaired and checked by reading, not compiled, not run on hardware
//
#include <hip/hip_runtime.h>
#include <hip/hip_bf16.h>
#include <cstdint>
#include <cstddef>

// ---------------------------------------------------------------------------
// Types for CDNA5 WMMA (wave32): v_wmma_f32_16x16x32_bf16
// ---------------------------------------------------------------------------
typedef __bf16 bf16;
typedef __attribute__((ext_vector_type(16))) __bf16 v16bf;
typedef __attribute__((ext_vector_type(8)))  float  v8f;
typedef __attribute__((ext_vector_type(4)))  unsigned int v4u;
typedef __attribute__((ext_vector_type(8)))  int v8i;
typedef __attribute__((ext_vector_type(4)))  int v4i;

#define NEG_SLOPE 0.2f

// ---------------------------------------------------------------------------
// float atomic max via sign-split int/uint atomics (correct for all floats,
// including the -1e30 init; mixed-sign races resolve correctly because
// positive floats order as small uints / positive ints).
// ---------------------------------------------------------------------------
static __device__ __forceinline__ void atomic_max_f32(float* addr, float val) {
  if (val >= 0.0f) {
    atomicMax((int*)addr, __float_as_int(val));
  } else {
    atomicMin((unsigned int*)addr, __float_as_uint(val));
  }
}

// edge i in [0, E): real edge from edge_index; edge i in [E, E+N): self loop.
static __device__ __forceinline__ void edge_ends(const int* __restrict__ ei,
                                                 int E, int e, int& s, int& d) {
  if (e < E) { s = ei[e]; d = ei[E + e]; }
  else       { s = e - E; d = e - E; }
}

// ---------------------------------------------------------------------------
// Conversions
// ---------------------------------------------------------------------------
__global__ void k_conv_x_bf16(const float* __restrict__ x, bf16* __restrict__ xb,
                              int N, int K, int Kpad) {
  int t = blockIdx.x * blockDim.x + threadIdx.x;
  if (t >= N * Kpad) return;
  int n = t / Kpad, k = t % Kpad;
  xb[t] = (k < K) ? (bf16)x[(size_t)n * K + k] : (bf16)0.0f;
}

// W [K x Nout] row-major (fp32) -> Wt [Nout x Kpad] bf16 (zero-padded K)
__global__ void k_conv_w_bf16_t(const float* __restrict__ W, bf16* __restrict__ Wt,
                                int K, int Kpad, int Nout) {
  int t = blockIdx.x * blockDim.x + threadIdx.x;
  if (t >= Nout * Kpad) return;
  int n = t / Kpad, k = t % Kpad;
  Wt[t] = (k < K) ? (bf16)W[(size_t)k * Nout + n] : (bf16)0.0f;
}

__global__ void k_fill_f32(float* __restrict__ p, float v, int n) {
  int t = blockIdx.x * blockDim.x + threadIdx.x;
  if (t < n) p[t] = v;
}

// ---------------------------------------------------------------------------
// WMMA GEMM with TDM-staged A panel:
//   D[M x Nout] (bf16) = A[M x Kpad] (bf16, row-major) @ Bt[Nout x Kpad] (bf16,
//   pre-transposed, N-major so each lane's B fragment is contiguous).
// One block (4 waves) per 16-row M-tile:
//   - wave 0 DMAs the 16 x Kpad bf16 A panel into LDS via tensor_load_to_lds
//     (D# built per CDNA5 ISA 8.3/8.4), waits TENSORcnt==0, then barrier.
//   - each wave preloads all KSTEPS A fragments into registers from LDS and
//     iterates over its share of N-tiles, so A global traffic is paid once.
// Requires M % 16 == 0, Nout % 16 == 0 (true here).
// ---------------------------------------------------------------------------
template <int KSTEPS>
__global__ void k_wmma_gemm_lds(const bf16* __restrict__ A,
                                const bf16* __restrict__ Bt,
                                bf16* __restrict__ D,
                                int M, int Nout) {
  constexpr int Kpad = KSTEPS * 32;
  extern __shared__ bf16 Atile[];          // 16 x Kpad, LDS offset 0

  const int m0 = blockIdx.x << 4;

#if __has_builtin(__builtin_amdgcn_tensor_load_to_lds)
  if (threadIdx.x < 32) {                  // wave 0 issues the TDM transfer
    unsigned long long ga = (unsigned long long)(A + (size_t)m0 * Kpad);
    // D# group0: count=1 | lds_addr=0 | global_addr[56:0] | type=2 (bits 127:126)
    v4u g0 = { 1u,
               0u,
               (unsigned)(ga & 0xffffffffu),
               (unsigned)((ga >> 32) & 0x01ffffffu) | 0x80000000u };
    // D# group1: data_size=2B; tensor_dim0=Kpad; tensor_dim1=16;
    //            tile_dim0=Kpad; tile_dim1=16; tensor_dim0_stride=Kpad
    v8i g1 = { (int)(1u << 16),                       // [17:16] data_size=1 (2B)
               (int)((unsigned)Kpad << 16),           // [63:48] tensor_dim0 lo16
               (int)(16u << 16),                      // [79:64]=dim0 hi16(0), [95:80]=tensor_dim1 lo16
               (int)((unsigned)Kpad << 16),           // [111:96]=dim1 hi16(0), [127:112]=tile_dim0
               16,                                    // [143:128]=tile_dim1, [159:144]=tile_dim2(0)
               Kpad,                                  // [191:160] tensor_dim0_stride lo32
               0, 0 };                                // stride hi / tensor_dim1_stride = 0
    v4i gz4 = { 0, 0, 0, 0 };
    v8i gz8 = { 0, 0, 0, 0, 0, 0, 0, 0 };
    // 6-arg toolchain form: (g0, g1, g2, g3, extra, cpol)
    __builtin_amdgcn_tensor_load_to_lds(g0, g1, gz4, gz4, gz8, 0);
    __builtin_amdgcn_s_wait_tensorcnt(0);
  }
  __syncthreads();
#else
  for (int i = threadIdx.x; i < (16 * Kpad * (int)sizeof(bf16)) / 16; i += blockDim.x) {
    ((uint4*)Atile)[i] = ((const uint4*)(A + (size_t)m0 * Kpad))[i];
  }
  __syncthreads();
#endif

  const int lane = threadIdx.x & 31;
  const int wave = threadIdx.x >> 5;
  const int wpb  = blockDim.x >> 5;
  const int h    = lane >> 4;              // lane half (0/1)
  const int l    = lane & 15;              // row (A) / col (B, C/D)
  const int Ntiles = Nout >> 4;

  // Preload A fragments for every K-step from LDS.
  // A fragment 16x32 bf16: VGPR v (pair 2v,2v+1): K = 16*(v>>2) + 8*h + 2*(v&3)
  v16bf areg[KSTEPS];
#pragma unroll
  for (int kk = 0; kk < KSTEPS; ++kk) {
    const bf16* Arow = Atile + l * Kpad + kk * 32 + 8 * h;
#pragma unroll
    for (int v = 0; v < 8; ++v) {
      int koff = ((v >> 2) << 4) + ((v & 3) << 1);
      areg[kk][2 * v]     = Arow[koff];
      areg[kk][2 * v + 1] = Arow[koff + 1];
    }
  }

  for (int nt = wave; nt < Ntiles; nt += wpb) {   // wave-uniform loop
    const int n0 = nt << 4;
    v8f acc = {};
#pragma unroll
    for (int kk = 0; kk < KSTEPS; ++kk) {
      // B fragment 32x16 bf16: VGPR j: K = 16*(j>>2) + 8*h + 2*(j&3), +1 ; N = l.
      v16bf b;
      const bf16* Bcol = Bt + (size_t)(n0 + l) * Kpad + kk * 32 + 8 * h;
#pragma unroll
      for (int j = 0; j < 8; ++j) {
        int koff = ((j >> 2) << 4) + ((j & 3) << 1);
        b[2 * j]     = Bcol[koff];
        b[2 * j + 1] = Bcol[koff + 1];
      }
      acc = __builtin_amdgcn_wmma_f32_16x16x32_bf16(
          /*neg_a=*/false, areg[kk], /*neg_b=*/false, b,
          /*c_mod=*/(short)0, acc, /*reuse_a=*/false, /*reuse_b=*/false);
    }
    // C/D layout: VGPR r holds row (r + 8*h), col l
#pragma unroll
    for (int r = 0; r < 8; ++r) {
      D[(size_t)(m0 + r + 8 * h) * Nout + n0 + l] = (bf16)acc[r];
    }
  }
}

// ---------------------------------------------------------------------------
// Per-(node, head) attention scores: a_s = <h, att_src>, a_d = <h, att_dst>
// ---------------------------------------------------------------------------
__global__ void k_attn_scores(const bf16* __restrict__ Hm,
                              const float* __restrict__ att_src,
                              const float* __restrict__ att_dst,
                              float* __restrict__ a_s, float* __restrict__ a_d,
                              int N, int H, int C) {
  int t = blockIdx.x * blockDim.x + threadIdx.x;
  if (t >= N * H) return;
  int n = t / H, h = t % H;
  const bf16* row = Hm + (size_t)n * H * C + (size_t)h * C;
  float ss = 0.0f, sd = 0.0f;
  for (int c = 0; c < C; ++c) {
    float v = (float)row[c];
    ss += v * att_src[h * C + c];
    sd += v * att_dst[h * C + c];
  }
  a_s[t] = ss;
  a_d[t] = sd;
}

// ---------------------------------------------------------------------------
// Edge passes (one thread per (edge, head))
// ---------------------------------------------------------------------------
__global__ void k_edge_max(const int* __restrict__ ei, int E, int N, int H,
                           const float* __restrict__ a_s,
                           const float* __restrict__ a_d,
                           float* __restrict__ m) {
  int t = blockIdx.x * blockDim.x + threadIdx.x;
  if (t >= (E + N) * H) return;
  int e = t / H, h = t % H, s, d;
  edge_ends(ei, E, e, s, d);
  float v = a_s[s * H + h] + a_d[d * H + h];
  v = (v > 0.0f) ? v : NEG_SLOPE * v;
  atomic_max_f32(&m[d * H + h], v);
}

__global__ void k_edge_denom(const int* __restrict__ ei, int E, int N, int H,
                             const float* __restrict__ a_s,
                             const float* __restrict__ a_d,
                             const float* __restrict__ m,
                             float* __restrict__ den) {
  int t = blockIdx.x * blockDim.x + threadIdx.x;
  if (t >= (E + N) * H) return;
  int e = t / H, h = t % H, s, d;
  edge_ends(ei, E, e, s, d);
  float v = a_s[s * H + h] + a_d[d * H + h];
  v = (v > 0.0f) ? v : NEG_SLOPE * v;
  atomicAdd(&den[d * H + h], __expf(v - m[d * H + h]));
}

__global__ void k_edge_aggregate(const int* __restrict__ ei, int E, int N, int H, int C,
                                 const float* __restrict__ a_s,
                                 const float* __restrict__ a_d,
                                 const float* __restrict__ m,
                                 const float* __restrict__ den,
                                 const bf16* __restrict__ Hm,
                                 float* __restrict__ acc) {
  int t = blockIdx.x * blockDim.x + threadIdx.x;
  if (t >= (E + N) * H) return;
  int e = t / H, h = t % H, s, d;
  edge_ends(ei, E, e, s, d);
  float v = a_s[s * H + h] + a_d[d * H + h];
  v = (v > 0.0f) ? v : NEG_SLOPE * v;
  float alpha = __expf(v - m[d * H + h]) / den[d * H + h];
  const bf16* hr = Hm + (size_t)s * H * C + (size_t)h * C;
  float* ar = acc + (size_t)d * H * C + (size_t)h * C;
  for (int c = 0; c < C; ++c) {
    atomicAdd(&ar[c], alpha * (float)hr[c]);
  }
}

// ---------------------------------------------------------------------------
// acc = elu(acc + bias); optional bf16 copy for the next layer's GEMM input.
// ---------------------------------------------------------------------------
__global__ void k_finalize_elu(float* __restrict__ acc, const float* __restrict__ bias,
                               bf16* __restrict__ outbf, int N, int F) {
  int t = blockIdx.x * blockDim.x + threadIdx.x;
  if (t >= N * F) return;
  int f = t % F;
  float v = acc[t] + bias[f];
  v = (v > 0.0f) ? v : (__expf(v) - 1.0f);
  acc[t] = v;
  if (outbf) outbf[t] = (bf16)v;
}

// ---------------------------------------------------------------------------
// Link predictor: y = sigmoid(concat(x[n1], x[n2]) @ linW + linb)
// ---------------------------------------------------------------------------
__global__ void k_pair_predict(const float* __restrict__ x,
                               const int* __restrict__ n1,
                               const int* __restrict__ n2,
                               const float* __restrict__ linW,
                               const float* __restrict__ linb,
                               float* __restrict__ y, int P, int F) {
  int p = blockIdx.x * blockDim.x + threadIdx.x;
  if (p >= P) return;
  const float* r1 = x + (size_t)n1[p] * F;
  const float* r2 = x + (size_t)n2[p] * F;
  float acc0 = linb[0], acc1 = linb[1];
  for (int j = 0; j < F; ++j) {
    float v = r1[j];
    acc0 += v * linW[j * 2 + 0];
    acc1 += v * linW[j * 2 + 1];
  }
  for (int j = 0; j < F; ++j) {
    float v = r2[j];
    acc0 += v * linW[(F + j) * 2 + 0];
    acc1 += v * linW[(F + j) * 2 + 1];
  }
  y[p * 2 + 0] = 1.0f / (1.0f + __expf(-acc0));
  y[p * 2 + 1] = 1.0f / (1.0f + __expf(-acc1));
}

// ---------------------------------------------------------------------------
// Host-side launch sequence
// ---------------------------------------------------------------------------
static inline int nblk(long long n, int b) { return (int)((n + b - 1) / b); }

extern "C" void kernel_launch(void* const* d_in, const int* in_sizes, int n_in,
                              void* d_out, int out_size, void* d_ws, size_t ws_size,
                              hipStream_t stream) {
  (void)n_in; (void)out_size; (void)ws_size;

  // Inputs (setup_inputs dict order)
  const float* features = (const float*)d_in[0];
  const int*   edge_idx = (const int*)d_in[1];
  const int*   node1    = (const int*)d_in[2];
  const int*   node2    = (const int*)d_in[3];
  const float* W1       = (const float*)d_in[4];
  const float* att_s1   = (const float*)d_in[5];
  const float* att_d1   = (const float*)d_in[6];
  const float* b1       = (const float*)d_in[7];
  const float* W2       = (const float*)d_in[8];
  const float* att_s2   = (const float*)d_in[9];
  const float* att_d2   = (const float*)d_in[10];
  const float* b2       = (const float*)d_in[11];
  const float* linW     = (const float*)d_in[12];
  const float* linb     = (const float*)d_in[13];

  const int F0 = 213;            // input features
  const int KP1 = 224;           // padded to multiple of 32 (7 K-steps)
  const int H1 = 12, C1 = 16, F1 = H1 * C1;   // 192
  const int H2 = 8,  C2 = 8,  F2 = H2 * C2;   // 64
  const int KP2 = F1;            // 192 (6 K-steps)

  const int N = in_sizes[0] / F0;      // 50000
  const int E = in_sizes[1] / 2;       // 800000
  const int P = in_sizes[2];           // 16384

  // ---- workspace layout (256B aligned carve-outs) ----
  char* ws = (char*)d_ws;
  size_t off = 0;
  auto carve = [&](size_t bytes) -> char* {
    off = (off + 255) & ~(size_t)255;
    char* p = ws + off;
    off += bytes;
    return p;
  };
  bf16*  h_bf  = (bf16*)carve((size_t)N * F1 * sizeof(bf16));   // layer1 h; reused for h2 (N*F2)
  bf16*  x_bf  = (bf16*)carve((size_t)N * KP1 * sizeof(bf16));  // features bf16; reused for x1_bf (N*F1)
  bf16*  w_bf  = (bf16*)carve((size_t)KP1 * F1 * sizeof(bf16)); // transposed weights (max size)
  float* a_s   = (float*)carve((size_t)N * H1 * sizeof(float));
  float* a_d   = (float*)carve((size_t)N * H1 * sizeof(float));
  float* mx    = (float*)carve((size_t)N * H1 * sizeof(float));
  float* den   = (float*)carve((size_t)N * H1 * sizeof(float));
  float* acc1  = (float*)carve((size_t)N * F1 * sizeof(float));

  float* y_out = (float*)d_out;            // [P, 2]
  float* x_out = y_out + (size_t)2 * P;    // [N, F2] — also layer-2 accumulator

  const int B = 256;

  // ================= Layer 1 =================
  k_conv_x_bf16<<<nblk((long long)N * KP1, B), B, 0, stream>>>(features, x_bf, N, F0, KP1);
  k_conv_w_bf16_t<<<nblk((long long)F1 * KP1, B), B, 0, stream>>>(W1, w_bf, F0, KP1, F1);

  k_wmma_gemm_lds<7><<<N >> 4, 128, 16 * KP1 * sizeof(bf16), stream>>>(x_bf, w_bf, h_bf, N, F1);

  k_attn_scores<<<nblk((long long)N * H1, B), B, 0, stream>>>(h_bf, att_s1, att_d1, a_s, a_d, N, H1, C1);

  k_fill_f32<<<nblk((long long)N * H1, B), B, 0, stream>>>(mx, -1e30f, N * H1);
  k_fill_f32<<<nblk((long long)N * H1, B), B, 0, stream>>>(den, 0.0f, N * H1);
  k_fill_f32<<<nblk((long long)N * F1, B), B, 0, stream>>>(acc1, 0.0f, N * F1);

  {
    long long nt = (long long)(E + N) * H1;
    k_edge_max<<<nblk(nt, B), B, 0, stream>>>(edge_idx, E, N, H1, a_s, a_d, mx);
    k_edge_denom<<<nblk(nt, B), B, 0, stream>>>(edge_idx, E, N, H1, a_s, a_d, mx, den);
    k_edge_aggregate<<<nblk(nt, B), B, 0, stream>>>(edge_idx, E, N, H1, C1, a_s, a_d, mx, den, h_bf, acc1);
  }

  // acc1 = elu(acc1 + b1); x1_bf (reuses x_bf region, [N x 192])
  k_finalize_elu<<<nblk((long long)N * F1, B), B, 0, stream>>>(acc1, b1, x_bf, N, F1);

  // ================= Layer 2 =================
  k_conv_w_bf16_t<<<nblk((long long)F2 * KP2, B), B, 0, stream>>>(W2, w_bf, F1, KP2, F2);

  k_wmma_gemm_lds<6><<<N >> 4, 128, 16 * KP2 * sizeof(bf16), stream>>>(x_bf, w_bf, h_bf, N, F2);

  k_attn_scores<<<nblk((long long)N * H2, B), B, 0, stream>>>(h_bf, att_s2, att_d2, a_s, a_d, N, H2, C2);

  k_fill_f32<<<nblk((long long)N * H2, B), B, 0, stream>>>(mx, -1e30f, N * H2);
  k_fill_f32<<<nblk((long long)N * H2, B), B, 0, stream>>>(den, 0.0f, N * H2);
  k_fill_f32<<<nblk((long long)N * F2, B), B, 0, stream>>>(x_out, 0.0f, N * F2);

  {
    long long nt = (long long)(E + N) * H2;
    k_edge_max<<<nblk(nt, B), B, 0, stream>>>(edge_idx, E, N, H2, a_s, a_d, mx);
    k_edge_denom<<<nblk(nt, B), B, 0, stream>>>(edge_idx, E, N, H2, a_s, a_d, mx, den);
    k_edge_aggregate<<<nblk(nt, B), B, 0, stream>>>(edge_idx, E, N, H2, C2, a_s, a_d, mx, den, h_bf, x_out);
  }

  // x_out = elu(x_out + b2) in place (this is the second tuple output)
  k_finalize_elu<<<nblk((long long)N * F2, B), B, 0, stream>>>(x_out, b2, (bf16*)nullptr, N, F2);

  // y_pred
  k_pair_predict<<<nblk(P, B), B, 0, stream>>>(x_out, node1, node2, linW, linb, y_out, P, F2);
}
